// Attention_9947144257674
// MI455X (gfx1250) — matrix operand order repaired and checked by reading, not compiled, not run on hardware
//
#include <hip/hip_runtime.h>
#include <hip/hip_bf16.h>

// ---------------------------------------------------------------------------
// CDNA5 (gfx1250) causal multi-head attention, f16 WMMA with f32 accumulate.
//   B=4, S=2048, N_HEADS=16, D_HEAD=64, D_MODEL=1024
// ---------------------------------------------------------------------------

typedef _Float16 v16h __attribute__((ext_vector_type(16)));
typedef _Float16 v8h  __attribute__((ext_vector_type(8)));
typedef _Float16 v4h  __attribute__((ext_vector_type(4)));
typedef float    v8f  __attribute__((ext_vector_type(8)));

#define DM 1024   // d_model
#define DH 64     // d_head
#define NH 16     // heads
#define SQ 2048   // seq len
#define NB 4      // batch
#define EPSILON (-100000.0f)

static __device__ __forceinline__ v8f wmma32(v16h a, v16h b, v8f c) {
  // v_wmma_f32_16x16x32_f16: D = A(16x32 f16) x B(32x16 f16) + C(16x16 f32)
  return __builtin_amdgcn_wmma_f32_16x16x32_f16(false, a, false, b,
                                                (short)0, c, false, false);
}

// A-matrix (16x32 f16) lane fragment. Lane: m = lane&15, hf = lane>>4.
// VGPR i holds K pair k0 = (i/4)*16 + hf*8 + (i%4)*2  (ISA 7.12.2).
// => elements 0..7  = row[hf*8   .. hf*8+7]
//    elements 8..15 = row[16+hf*8 .. 16+hf*8+7]     (two 16B loads)
static __device__ __forceinline__ v16h load_a(const _Float16* __restrict__ rowp, int hf) {
  v8h lo = *(const v8h*)(rowp + hf * 8);
  v8h hi = *(const v8h*)(rowp + 16 + hf * 8);
  return __builtin_shufflevector(lo, hi, 0,1,2,3,4,5,6,7,8,9,10,11,12,13,14,15);
}

// B-matrix (32x16 f16) lane fragment. Lane: n = lane&15, hf = lane>>4.
// Same K order as load_a: elements 0..7 = row[hf*8..], 8..15 = row[16+hf*8..].
static __device__ __forceinline__ v16h load_b(const _Float16* __restrict__ rowp, int hf) {
  v8h lo = *(const v8h*)(rowp + hf * 8);
  v8h hi = *(const v8h*)(rowp + 16 + hf * 8);
  return __builtin_shufflevector(lo, hi, 0,1,2,3,4,5,6,7,8,9,10,11,12,13,14,15);
}

// ------------------------------- converters --------------------------------

__global__ void __launch_bounds__(256) cvt_x_kernel(const float* __restrict__ in,
                                                    _Float16* __restrict__ out, int n) {
  int i = (blockIdx.x * 256 + threadIdx.x) * 4;
  if (i + 3 < n) {
    float4 v = *(const float4*)(in + i);
    v4h h; h[0] = (_Float16)v.x; h[1] = (_Float16)v.y;
    h[2] = (_Float16)v.z; h[3] = (_Float16)v.w;
    *(v4h*)(out + i) = h;
  }
}

// W_{Q,K,V}[n,e,h] (f32) -> w2[(n*64+h)*1024 + e] (f16)  (pairs contiguous in e)
__global__ void __launch_bounds__(256) cvt_wqkv_kernel(const float* __restrict__ in,
                                                       _Float16* __restrict__ out) {
  int t = blockIdx.x * 256 + threadIdx.x;      // 0 .. 1M-1
  int col = t >> 10, e = t & 1023;
  int n = col >> 6, h = col & 63;
  out[t] = (_Float16)in[(n << 16) + (e << 6) + h];
}

// W_O[n,h,e] (f32) -> who[e*1024 + (n*64+h)] (f16)  (pairs contiguous in nh)
__global__ void __launch_bounds__(256) cvt_wo_kernel(const float* __restrict__ in,
                                                     _Float16* __restrict__ out) {
  int t = blockIdx.x * 256 + threadIdx.x;      // 0 .. 1M-1
  int e = t >> 10, nh = t & 1023;
  out[t] = (_Float16)in[(nh << 10) + e];
}

// ------------------------------- QKV GEMM ----------------------------------
// One wave computes a 32x64 strip (two 16-row M tiles share all B fragments):
//   C[32 x 64] = X[32 x 1024] * W2^T[1024 x 64]
// 8 WMMAs per 32-wide K step vs 12 16B loads -> 1.5 loads/WMMA.
// grid = (256/4, NH, 3), block = 128 (4 waves).
__global__ void __launch_bounds__(128) qkv_gemm_kernel(
    const _Float16* __restrict__ xh,
    const _Float16* __restrict__ w2q, const _Float16* __restrict__ w2k,
    const _Float16* __restrict__ w2v,
    const float* __restrict__ bq, const float* __restrict__ bk,
    const float* __restrict__ bv,
    _Float16* __restrict__ qo, _Float16* __restrict__ ko,
    _Float16* __restrict__ vto) {
  const int lane = threadIdx.x & 31;
  const int wave = threadIdx.x >> 5;
  const int mtile = blockIdx.x * 4 + wave;     // 0..255 (32-row tiles over B*S)
  const int head = blockIdx.y;                 // 0..15
  const int which = blockIdx.z;                // 0=Q 1=K 2=V

  const _Float16* w2 = (which == 0) ? w2q : (which == 1) ? w2k : w2v;
  const float* bias = (which == 0) ? bq : (which == 1) ? bk : bv;

  const int cl = lane & 15;                    // B/C column, A row
  const int hf = lane >> 4;

  const _Float16* arow0 = xh + (size_t)(mtile * 32 + cl) * DM;
  const _Float16* arow1 = xh + (size_t)(mtile * 32 + 16 + cl) * DM;
  const size_t wrow0 = (size_t)(head * 64 + 0 * 16 + cl) * DM;
  const size_t wrow1 = (size_t)(head * 64 + 1 * 16 + cl) * DM;
  const size_t wrow2 = (size_t)(head * 64 + 2 * 16 + cl) * DM;
  const size_t wrow3 = (size_t)(head * 64 + 3 * 16 + cl) * DM;

  v8f c00 = {}, c01 = {}, c02 = {}, c03 = {};  // M-tile 0
  v8f c10 = {}, c11 = {}, c12 = {}, c13 = {};  // M-tile 1
#pragma unroll 2
  for (int kbk = 0; kbk < DM; kbk += 32) {
    // prefetch next K block (global_prefetch_b8)
    if (kbk + 32 < DM) {
      __builtin_prefetch(arow0 + kbk + 32, 0, 3);
      __builtin_prefetch(arow1 + kbk + 32, 0, 3);
      __builtin_prefetch(w2 + wrow0 + kbk + 32, 0, 3);
    }
    v16h a0 = load_a(arow0 + kbk, hf);
    v16h a1 = load_a(arow1 + kbk, hf);
    v16h b0 = load_b(w2 + wrow0 + kbk, hf);
    v16h b1 = load_b(w2 + wrow1 + kbk, hf);
    v16h b2 = load_b(w2 + wrow2 + kbk, hf);
    v16h b3 = load_b(w2 + wrow3 + kbk, hf);
    c00 = wmma32(a0, b0, c00); c10 = wmma32(a1, b0, c10);
    c01 = wmma32(a0, b1, c01); c11 = wmma32(a1, b1, c11);
    c02 = wmma32(a0, b2, c02); c12 = wmma32(a1, b2, c12);
    c03 = wmma32(a0, b3, c03); c13 = wmma32(a1, b3, c13);
  }
  v8f cacc[2][4] = {{c00, c01, c02, c03}, {c10, c11, c12, c13}};

  const int bidx = mtile >> 6;                 // 64 32-row tiles per batch entry
  const int s0 = (mtile & 63) * 32;

  if (which < 2) {
    // q/k: [b, n, s, h] row-major (h contiguous)
    _Float16* outp = ((which == 0) ? qo : ko) +
                     (size_t)(bidx * NH + head) * SQ * DH;
#pragma unroll
    for (int mt = 0; mt < 2; ++mt) {
#pragma unroll
      for (int sub = 0; sub < 4; ++sub) {
        float bb = bias[head * 64 + sub * 16 + cl];
#pragma unroll
        for (int r = 0; r < 8; ++r) {
          int row = s0 + mt * 16 + hf * 8 + r; // C layout: row = hf*8 + r
          outp[(size_t)row * DH + sub * 16 + cl] =
              (_Float16)(cacc[mt][sub][r] + bb);
        }
      }
    }
  } else {
    // v: transposed [b, n, h, s] so PV B-operand loads are contiguous in s
    _Float16* outp = vto + (size_t)(bidx * NH + head) * DH * SQ;
#pragma unroll
    for (int mt = 0; mt < 2; ++mt) {
#pragma unroll
      for (int sub = 0; sub < 4; ++sub) {
        float bb = bias[head * 64 + sub * 16 + cl];
        v8h pk;
#pragma unroll
        for (int r = 0; r < 8; ++r) pk[r] = (_Float16)(cacc[mt][sub][r] + bb);
        *(v8h*)(outp + (size_t)(sub * 16 + cl) * SQ + s0 + mt * 16 + hf * 8) = pk;
      }
    }
  }
}

// ---------------------------- flash attention ------------------------------
// One wave per (b, head, 16-query tile). Keys processed in 32-wide chunks
// with online softmax. grid = (S/16, NB*NH), block = 32.
__global__ void __launch_bounds__(32) attn_kernel(
    const _Float16* __restrict__ q, const _Float16* __restrict__ k,
    const _Float16* __restrict__ vt, _Float16* __restrict__ z) {
  __shared__ float lds[16][36];                // 16 rows x 32 probs (+pad)

  const int lane = threadIdx.x & 31;
  const int cl = lane & 15;
  const int hf = lane >> 4;
  const int qtile = blockIdx.x;                // 0..127
  const int bn = blockIdx.y;                   // 0..63
  const int b = bn >> 4, hd = bn & 15;
  const int q0 = qtile * 16;

  const _Float16* qp = q + (size_t)bn * SQ * DH;
  const _Float16* kp = k + (size_t)bn * SQ * DH;
  const _Float16* vp = vt + (size_t)bn * DH * SQ;

  // Q fragments (A-operand), e in [0,32) and [32,64)
  const v16h aq0 = load_a(qp + (size_t)(q0 + cl) * DH + 0, hf);
  const v16h aq1 = load_a(qp + (size_t)(q0 + cl) * DH + 32, hf);

  v8f o0 = {}, o1 = {}, o2 = {}, o3 = {};
  float M[8], L[8];
#pragma unroll
  for (int r = 0; r < 8; ++r) { M[r] = -1e30f; L[r] = 0.0f; }

  const int nch = (q0 + 16 + 31) >> 5;         // chunks of 32 keys (causal)
  for (int ch = 0; ch < nch; ++ch) {
    const int kc = ch * 32;

    // ---- scores: S = Q x K^T  (two 16-key tiles) ----
    v8f s0 = {}, s1 = {};
    s0 = wmma32(aq0, load_b(kp + (size_t)(kc + cl) * DH + 0, hf), s0);
    s1 = wmma32(aq0, load_b(kp + (size_t)(kc + 16 + cl) * DH + 0, hf), s1);
    s0 = wmma32(aq1, load_b(kp + (size_t)(kc + cl) * DH + 32, hf), s0);
    s1 = wmma32(aq1, load_b(kp + (size_t)(kc + 16 + cl) * DH + 32, hf), s1);

    // ---- scale + causal mask (reference semantics: masked = -1e5) ----
#pragma unroll
    for (int r = 0; r < 8; ++r) {
      const int row = q0 + hf * 8 + r;
      float x0 = s0[r] * 0.125f;               // 1/sqrt(64)
      float x1 = s1[r] * 0.125f;
      if (kc + cl > row) x0 = EPSILON;
      if (kc + 16 + cl > row) x1 = EPSILON;
      s0[r] = x0; s1[r] = x1;
    }

    // ---- online softmax (row reductions across 16 lanes of each half) ----
    float al[8];
#pragma unroll
    for (int r = 0; r < 8; ++r) {
      float mx = fmaxf(s0[r], s1[r]);
#pragma unroll
      for (int off = 8; off > 0; off >>= 1) mx = fmaxf(mx, __shfl_xor(mx, off, 16));
      const float nm = fmaxf(M[r], mx);
      al[r] = __expf(M[r] - nm);
      M[r] = nm;
    }
#pragma unroll
    for (int r = 0; r < 8; ++r) {
      const float p0 = __expf(s0[r] - M[r]);
      const float p1 = __expf(s1[r] - M[r]);
      float rs = p0 + p1;
#pragma unroll
      for (int off = 8; off > 0; off >>= 1) rs += __shfl_xor(rs, off, 16);
      L[r] = L[r] * al[r] + rs;
      o0[r] *= al[r]; o1[r] *= al[r]; o2[r] *= al[r]; o3[r] *= al[r];
      lds[hf * 8 + r][cl] = p0;                // C-layout -> LDS
      lds[hf * 8 + r][16 + cl] = p1;
    }
    __syncthreads();

    // ---- reshape probs: C-layout -> A-layout (16x32 f16) ----
    v16h ap;
#pragma unroll
    for (int i = 0; i < 8; ++i) {
      const int k0 = ((i >> 2) << 4) + hf * 8 + ((i & 3) << 1);
      const float2 pv = *(const float2*)&lds[cl][k0];
      ap[2 * i] = (_Float16)pv.x;
      ap[2 * i + 1] = (_Float16)pv.y;
    }
    __syncthreads();

    // ---- O += P x V  (V transposed: contiguous along keys) ----
    o0 = wmma32(ap, load_b(vp + (size_t)(0 * 16 + cl) * SQ + kc, hf), o0);
    o1 = wmma32(ap, load_b(vp + (size_t)(1 * 16 + cl) * SQ + kc, hf), o1);
    o2 = wmma32(ap, load_b(vp + (size_t)(2 * 16 + cl) * SQ + kc, hf), o2);
    o3 = wmma32(ap, load_b(vp + (size_t)(3 * 16 + cl) * SQ + kc, hf), o3);
  }

  // ---- normalize and emit z[b, s, n*64+h] (f16) ----
  v8f oacc[4] = {o0, o1, o2, o3};
#pragma unroll
  for (int sub = 0; sub < 4; ++sub) {
#pragma unroll
    for (int r = 0; r < 8; ++r) {
      const int row = q0 + hf * 8 + r;
      z[(size_t)(b * SQ + row) * DM + hd * 64 + sub * 16 + cl] =
          (_Float16)(oacc[sub][r] / L[r]);
    }
  }
}

// --------------------------- output projection -----------------------------
// out[b*s, e] = z[b*s, nh] * who[e, nh]^T + b_O.
// 32x64 strip per wave (two M tiles share B). grid = (64, 16), block 128.
__global__ void __launch_bounds__(128) oproj_kernel(
    const _Float16* __restrict__ zh, const _Float16* __restrict__ who,
    const float* __restrict__ bO, float* __restrict__ out) {
  const int lane = threadIdx.x & 31;
  const int wave = threadIdx.x >> 5;
  const int mtile = blockIdx.x * 4 + wave;     // 0..255 (32-row tiles)
  const int cstrip = blockIdx.y;               // 64-col strip

  const int cl = lane & 15;
  const int hf = lane >> 4;

  const _Float16* arow0 = zh + (size_t)(mtile * 32 + cl) * DM;
  const _Float16* arow1 = zh + (size_t)(mtile * 32 + 16 + cl) * DM;
  const size_t wrow0 = (size_t)(cstrip * 64 + 0 * 16 + cl) * DM;
  const size_t wrow1 = (size_t)(cstrip * 64 + 1 * 16 + cl) * DM;
  const size_t wrow2 = (size_t)(cstrip * 64 + 2 * 16 + cl) * DM;
  const size_t wrow3 = (size_t)(cstrip * 64 + 3 * 16 + cl) * DM;

  v8f c00 = {}, c01 = {}, c02 = {}, c03 = {};
  v8f c10 = {}, c11 = {}, c12 = {}, c13 = {};
#pragma unroll 2
  for (int kbk = 0; kbk < DM; kbk += 32) {
    if (kbk + 32 < DM) {
      __builtin_prefetch(arow0 + kbk + 32, 0, 3);
      __builtin_prefetch(arow1 + kbk + 32, 0, 3);
      __builtin_prefetch(who + wrow0 + kbk + 32, 0, 3);
    }
    v16h a0 = load_a(arow0 + kbk, hf);
    v16h a1 = load_a(arow1 + kbk, hf);
    v16h b0 = load_b(who + wrow0 + kbk, hf);
    v16h b1 = load_b(who + wrow1 + kbk, hf);
    v16h b2 = load_b(who + wrow2 + kbk, hf);
    v16h b3 = load_b(who + wrow3 + kbk, hf);
    c00 = wmma32(a0, b0, c00); c10 = wmma32(a1, b0, c10);
    c01 = wmma32(a0, b1, c01); c11 = wmma32(a1, b1, c11);
    c02 = wmma32(a0, b2, c02); c12 = wmma32(a1, b2, c12);
    c03 = wmma32(a0, b3, c03); c13 = wmma32(a1, b3, c13);
  }
  v8f cacc[2][4] = {{c00, c01, c02, c03}, {c10, c11, c12, c13}};

#pragma unroll
  for (int mt = 0; mt < 2; ++mt) {
#pragma unroll
    for (int sub = 0; sub < 4; ++sub) {
      const int col = cstrip * 64 + sub * 16 + cl;
      const float bb = bO[col];
#pragma unroll
      for (int r = 0; r < 8; ++r) {
        out[(size_t)(mtile * 32 + mt * 16 + hf * 8 + r) * DM + col] =
            cacc[mt][sub][r] + bb;
      }
    }
  }
}

// -------------------------------- launcher ---------------------------------

extern "C" void kernel_launch(void* const* d_in, const int* in_sizes, int n_in,
                              void* d_out, int out_size, void* d_ws, size_t ws_size,
                              hipStream_t stream) {
  const float* x = (const float*)d_in[0];     // [4,2048,1024]
  const float* W_Q = (const float*)d_in[1];   // [16,1024,64]
  const float* b_Q = (const float*)d_in[2];   // [16,64]
  const float* W_K = (const float*)d_in[3];
  const float* b_K = (const float*)d_in[4];
  const float* W_V = (const float*)d_in[5];
  const float* b_V = (const float*)d_in[6];
  const float* W_O = (const float*)d_in[7];   // [16,64,1024]
  const float* b_O = (const float*)d_in[8];   // [1024]
  float* out = (float*)d_out;                 // [4,2048,1024]

  // workspace carve-up (f16 elements)
  _Float16* ws = (_Float16*)d_ws;
  const size_t NX = (size_t)NB * SQ * DM;     // 8388608
  const size_t NW = (size_t)DM * DM;          // 1048576
  _Float16* xh  = ws;
  _Float16* w2q = xh + NX;
  _Float16* w2k = w2q + NW;
  _Float16* w2v = w2k + NW;
  _Float16* who = w2v + NW;
  _Float16* qb  = who + NW;                   // [b,n,s,h]
  _Float16* kb  = qb + NX;                    // [b,n,s,h]
  _Float16* vtb = kb + NX;                    // [b,n,h,s]
  _Float16* zb  = vtb + NX;                   // [b,s,n*h]
  (void)ws_size; (void)in_sizes; (void)n_in; (void)out_size;

  // 1) precision conversion + weight transposes
  cvt_x_kernel<<<(int)(NX / (256 * 4)), 256, 0, stream>>>(x, xh, (int)NX);
  cvt_wqkv_kernel<<<(int)(NW / 256), 256, 0, stream>>>(W_Q, w2q);
  cvt_wqkv_kernel<<<(int)(NW / 256), 256, 0, stream>>>(W_K, w2k);
  cvt_wqkv_kernel<<<(int)(NW / 256), 256, 0, stream>>>(W_V, w2v);
  cvt_wo_kernel<<<(int)(NW / 256), 256, 0, stream>>>(W_O, who);

  // 2) fused QKV projection (WMMA, 32x64 tiles)
  qkv_gemm_kernel<<<dim3(64, NH, 3), 128, 0, stream>>>(
      xh, w2q, w2k, w2v, b_Q, b_K, b_V, qb, kb, vtb);

  // 3) causal flash attention (WMMA + online softmax)
  attn_kernel<<<dim3(SQ / 16, NB * NH), 32, 0, stream>>>(qb, kb, vtb, zb);

  // 4) output projection (WMMA, 32x64 tiles) -> fp32
  oproj_kernel<<<dim3(64, NH), 128, 0, stream>>>(zb, who, b_O, out);
}
